// TreeModel_57741540327707
// MI455X (gfx1250) — matrix operand order repaired and checked by reading, not compiled, not run on hardware
//
#include <hip/hip_runtime.h>

typedef __attribute__((ext_vector_type(16))) _Float16     v16h;
typedef __attribute__((ext_vector_type(4)))  _Float16     v4h;
typedef __attribute__((ext_vector_type(8)))  float        v8f;
typedef __attribute__((ext_vector_type(4)))  float        v4f;
typedef __attribute__((ext_vector_type(4)))  unsigned int v4u;
typedef __attribute__((ext_vector_type(8)))  int          v8i;
typedef __attribute__((ext_vector_type(4)))  int          v4i;

#define B_ROWS  16384
#define F_DIM   1024   // feature dim (K for GEMM1)
#define N_INNER 1023   // inner tree nodes (2^10 - 1)
#define N_LEAF  1024   // leaves (K for GEMM2)
#define N_CLS   1000   // classes
#define M_TILE  64     // rows per block (4 WMMA m-subtiles)
#define NPAD    1024   // padded column count for transposed f16 weights

// ---------------------------------------------------------------------------
// Prep 1: transpose + convert both weight matrices to f16 in workspace,
// zero-padded to 1024 columns so the main kernel needs no bounds branches.
// ---------------------------------------------------------------------------
__global__ __launch_bounds__(256) void prep_weights_kernel(
    const float* __restrict__ Wg, const float* __restrict__ leaf,
    _Float16* __restrict__ WgT, _Float16* __restrict__ leafT)
{
    const int idx  = blockIdx.x * 256 + threadIdx.x;   // 0 .. 2*1024*1024-1
    const int half = idx >> 20;
    const int e    = idx & (NPAD * NPAD - 1);
    const int n    = e >> 10;
    const int k    = e & (NPAD - 1);
    if (half == 0) {
        const float v = (n < N_INNER) ? Wg[(size_t)k * N_INNER + n] : 0.f;
        WgT[e] = (_Float16)v;
    } else {
        const float v = (n < N_CLS) ? leaf[(size_t)k * N_CLS + n] : 0.f;
        leafT[e] = (_Float16)v;
    }
}

// ---------------------------------------------------------------------------
// Prep 2: convert x (16384x1024 f32) to f16 in workspace (coalesced both ways)
// so the main kernel can TDM the f16 tile straight into LDS.
// ---------------------------------------------------------------------------
__global__ __launch_bounds__(256) void prep_x_kernel(
    const float* __restrict__ x, _Float16* __restrict__ xh)
{
    const int i = blockIdx.x * 256 + threadIdx.x;   // over float4s
    const v4f v = ((const v4f*)x)[i];
    v4h h;
    h[0] = (_Float16)v.x;
    h[1] = (_Float16)v.y;
    h[2] = (_Float16)v.z;
    h[3] = (_Float16)v.w;
    ((v4h*)xh)[i] = h;
}

// ---------------------------------------------------------------------------
// Fused: p = sigmoid(x@Wg + b); tree-expand to leaf probs; out = probs@leaf.
// One block = 8 waves handles a 64-row tile; x tile arrives via one TDM DMA.
// ---------------------------------------------------------------------------
__global__ __launch_bounds__(256) void tree_fused_kernel(
    const _Float16* __restrict__ xh,
    const _Float16* __restrict__ WgT,
    const float* __restrict__ bg,
    const _Float16* __restrict__ leafT,
    float* __restrict__ out)
{
    extern __shared__ char smem[];
    _Float16* shA = (_Float16*)smem;             // 128 KB: x f16, later probs f16
    _Float16* shP = (_Float16*)(smem + 131072);  // 128 KB: gate probs p (f16)

    const int tid   = threadIdx.x;
    const int wave  = tid >> 5;
    const int lane  = tid & 31;
    const int lmod  = lane & 15;   // m for A, n for B/C
    const int lhalf = lane >> 4;   // which K-half this lane owns
    const int row0  = blockIdx.x * M_TILE;

    // ---- Stage 0: TDM load of the 64x1024 f16 x tile into LDS (wave 0) ----
    if (tid < 32) {
        const unsigned long long gaddr =
            (unsigned long long)(const void*)(xh + (size_t)row0 * F_DIM);
        const unsigned int ldsoff = (unsigned int)(unsigned long long)(void*)shA;
        v4u g0;
        g0[0] = 1u;                                        // count=1 valid descriptor
        g0[1] = ldsoff;                                    // lds_addr
        g0[2] = (unsigned int)(gaddr & 0xffffffffu);       // global_addr[31:0]
        g0[3] = (unsigned int)(gaddr >> 32) | (2u << 30);  // global_addr[56:32] | type=2
        v8i g1;
        g1[0] = (int)(1u << 16);                           // data_size = 2 bytes
        g1[1] = (int)((F_DIM & 0xffff) << 16);             // tensor_dim0[15:0]
        g1[2] = (int)((F_DIM >> 16) | ((B_ROWS & 0xffff) << 16)); // td0 hi | td1 lo
        g1[3] = (int)((B_ROWS >> 16) | (F_DIM << 16));     // td1 hi | tile_dim0 = 1024
        g1[4] = (int)M_TILE;                               // tile_dim1 = 64, tile_dim2 = 0
        g1[5] = (int)F_DIM;                                // tensor_dim0_stride[31:0]
        g1[6] = 0;                                         // stride0 hi | stride1 lo
        g1[7] = 0;
        v4i gz4 = {0, 0, 0, 0};
        v8i gz8 = {0, 0, 0, 0, 0, 0, 0, 0};
        __builtin_amdgcn_tensor_load_to_lds(g0, g1, gz4, gz4, gz8, 0);
        __builtin_amdgcn_s_wait_tensorcnt(0);
    }
    __syncthreads();

    // ---- Stage 1: p = sigmoid(x @ W_gate + b_gate) ----
    // 64 n-tiles of 16 cols, 8 per wave; each B fragment feeds 4 m-subtiles.
    for (int t = 0; t < 8; ++t) {
        const int ntile = wave * 8 + t;
        const int n     = ntile * 16 + lmod;
        const _Float16* brow = &WgT[(size_t)n << 10];
        v8f acc[4] = {};
#pragma unroll 2
        for (int k0 = 0; k0 < F_DIM; k0 += 32) {
            const v16h b  = *(const v16h*)(brow + k0 + lhalf * 16);
            const int kb0 = k0 + lhalf * 16;
            const int kb1 = k0 + lhalf * 16 + 8;
#pragma unroll
            for (int ms = 0; ms < 4; ++ms) {
                const _Float16* arow = &shA[(ms * 16 + lmod) * F_DIM];
                v16h a;
#pragma unroll
                for (int j = 0; j < 8; ++j) {
                    a[j]     = arow[kb0 + j];
                    a[8 + j] = arow[kb1 + j];
                }
                acc[ms] = __builtin_amdgcn_wmma_f32_16x16x32_f16(
                    false, a, false, b, (short)0, acc[ms], false, false);
            }
        }
        const float bias = (n < N_INNER) ? bg[n] : 0.f;
#pragma unroll
        for (int ms = 0; ms < 4; ++ms) {
#pragma unroll
            for (int i = 0; i < 8; ++i) {
                const int m = ms * 16 + i + lhalf * 8;
                const float z  = acc[ms][i] + bias;
                const float pv = 1.f / (1.f + __expf(-z));
                shP[m * N_LEAF + ntile * 16 + lmod] = (_Float16)pv;
            }
        }
    }
    __syncthreads();

    // ---- Stage 2: tree expansion -> probs (f16), overwriting shA ----
    // probs[m][L] = prod_l ( bit ? p[node] : 1-p[node] ),
    //   node = 2^l - 1 + (L >> (10-l)), bit = (L >> (9-l)) & 1
    for (int idx = tid; idx < M_TILE * N_LEAF; idx += 256) {
        const int m = idx >> 10;
        const int L = idx & (N_LEAF - 1);
        float prob = 1.f;
#pragma unroll
        for (int l = 0; l < 10; ++l) {
            const int node = (1 << l) - 1 + (L >> (10 - l));
            const float pv = (float)shP[m * N_LEAF + node];
            const int bit  = (L >> (9 - l)) & 1;
            prob *= bit ? pv : (1.f - pv);
        }
        shA[m * N_LEAF + L] = (_Float16)prob;
    }
    __syncthreads();

    // ---- Stage 3: out = probs @ leaf_logits ----
    for (int t = 0; t < 8; ++t) {
        const int ntile = wave * 8 + t;
        const int n0    = ntile * 16;
        if (n0 >= N_CLS) break;          // tile 63 (n0=1008) fully OOB
        const int n = n0 + lmod;
        const _Float16* brow = &leafT[(size_t)n << 10];
        v8f acc[4] = {};
#pragma unroll 2
        for (int k0 = 0; k0 < N_LEAF; k0 += 32) {
            const v16h b  = *(const v16h*)(brow + k0 + lhalf * 16);
            const int kb0 = k0 + lhalf * 16;
            const int kb1 = k0 + lhalf * 16 + 8;
#pragma unroll
            for (int ms = 0; ms < 4; ++ms) {
                const _Float16* arow = &shA[(ms * 16 + lmod) * N_LEAF];
                v16h a;
#pragma unroll
                for (int j = 0; j < 8; ++j) {
                    a[j]     = arow[kb0 + j];
                    a[8 + j] = arow[kb1 + j];
                }
                acc[ms] = __builtin_amdgcn_wmma_f32_16x16x32_f16(
                    false, a, false, b, (short)0, acc[ms], false, false);
            }
        }
        if (n < N_CLS) {
#pragma unroll
            for (int ms = 0; ms < 4; ++ms) {
#pragma unroll
                for (int i = 0; i < 8; ++i) {
                    const int m = ms * 16 + i + lhalf * 8;
                    out[(size_t)(row0 + m) * N_CLS + n] = acc[ms][i];
                }
            }
        }
    }
}

extern "C" void kernel_launch(void* const* d_in, const int* in_sizes, int n_in,
                              void* d_out, int out_size, void* d_ws, size_t ws_size,
                              hipStream_t stream) {
    (void)in_sizes; (void)n_in; (void)out_size; (void)ws_size;
    const float* x    = (const float*)d_in[0];
    const float* Wg   = (const float*)d_in[1];
    const float* bg   = (const float*)d_in[2];
    const float* leaf = (const float*)d_in[3];
    float* out        = (float*)d_out;

    _Float16* WgT   = (_Float16*)d_ws;                           // 2 MB
    _Float16* leafT = (_Float16*)d_ws + (size_t)NPAD * NPAD;     // 2 MB
    _Float16* xh    = (_Float16*)d_ws + (size_t)2 * NPAD * NPAD; // 32 MB

    prep_weights_kernel<<<(2 * NPAD * NPAD) / 256, 256, 0, stream>>>(
        Wg, leaf, WgT, leafT);
    prep_x_kernel<<<((size_t)B_ROWS * F_DIM / 4) / 256, 256, 0, stream>>>(x, xh);

    dim3 grid(B_ROWS / M_TILE);   // 256 blocks
    dim3 block(256);              // 8 waves of 32
    tree_fused_kernel<<<grid, block, 262144, stream>>>(xh, WgT, bg, leafT, out);
}
